// ResidualVectorQuantizer_29377576305407
// MI455X (gfx1250) — matrix operand order repaired and checked by reading, not compiled, not run on hardware
//
#include <hip/hip_runtime.h>
#include <stdint.h>

#define N_TOKENS   262144
#define EMB_DIM    128
#define NUM_EMB    1024
#define NUM_LEVELS 3
#define TM 64          // tokens per block
#define RS 136         // padded LDS row stride in bf16 elems (272B, conflict-free fragment loads)

typedef __attribute__((ext_vector_type(16))) __bf16 v16bf;
typedef __attribute__((ext_vector_type(8)))  float  v8f;

union V16BF { uint4 u[2]; v16bf v; };

__device__ __forceinline__ unsigned short f2bf(float f) {
  unsigned u = __float_as_uint(f);
  u = u + 0x7FFFu + ((u >> 16) & 1u);   // round-to-nearest-even
  return (unsigned short)(u >> 16);
}

// ---------------- Kernel 0: codebook f32 -> bf16 + per-code squared norms ----------------
__global__ void rvq_prep(const float* __restrict__ cb,
                         unsigned short* __restrict__ cb_bf,
                         float* __restrict__ norms) {
  __shared__ float red[128];
  const int row = blockIdx.x;          // 0 .. 3*1024-1
  const int t   = threadIdx.x;         // 0 .. 127
  float v = cb[(size_t)row * EMB_DIM + t];
  cb_bf[(size_t)row * EMB_DIM + t] = f2bf(v);
  red[t] = v * v;
  __syncthreads();
  for (int s = 64; s > 0; s >>= 1) { if (t < s) red[t] += red[t + s]; __syncthreads(); }
  if (t == 0) norms[row] = red[0];
}

// ---------------- Kernel 1: fused 3-level RVQ over 64-token tiles ----------------
__global__ __launch_bounds__(256)
void rvq_main(const float* __restrict__ z_e,
              const float* __restrict__ cb,                 // original fp32 codebooks
              const unsigned short* __restrict__ cb_bf,     // bf16 codebooks (L2-resident)
              const float* __restrict__ norms,
              float* __restrict__ out_zq,
              float* __restrict__ out_idx,
              float* __restrict__ partials) {
  __shared__ __align__(16) unsigned short r_bf[TM * RS];
  __shared__ __align__(16) unsigned short e_bf[TM * RS];
  __shared__ float scores[TM * 65];     // 64x64 tile, padded rows
  __shared__ float en_s[64];
  __shared__ float best[64];
  __shared__ int   bidx[64];
  __shared__ float red[256];

  const int t    = threadIdx.x;
  const int wave = t >> 5;
  const int lane = t & 31;
  const int tok  = t >> 2;              // local token 0..63 owned for load/update
  const int q    = t & 3;               // quarter of the 128 dims
  const size_t tokG0 = (size_t)blockIdx.x * TM;

  float rres[32];
  float zqt[32];
  {
    const float4* src = (const float4*)(z_e + (tokG0 + tok) * EMB_DIM + q * 32);
    #pragma unroll
    for (int i = 0; i < 8; ++i) {
      float4 v = src[i];
      rres[4*i+0] = v.x; rres[4*i+1] = v.y; rres[4*i+2] = v.z; rres[4*i+3] = v.w;
      zqt[4*i+0] = 0.f;  zqt[4*i+1] = 0.f;  zqt[4*i+2] = 0.f;  zqt[4*i+3] = 0.f;
    }
  }
  float sqacc = 0.0f;
  const unsigned lds_ebf = (unsigned)(uintptr_t)(&e_bf[0]);   // low 32 bits = LDS byte addr

  for (int lvl = 0; lvl < NUM_LEVELS; ++lvl) {
    // residual -> bf16 in LDS (packed pair stores)
    {
      unsigned* dst = (unsigned*)&r_bf[tok * RS + q * 32];
      #pragma unroll
      for (int i = 0; i < 16; ++i) {
        unsigned lo = f2bf(rres[2*i]);
        unsigned hi = f2bf(rres[2*i+1]);
        dst[i] = lo | (hi << 16);
      }
    }
    if (t < 64) { best[t] = 3.4e38f; bidx[t] = 0; }
    __syncthreads();

    const unsigned short* cbl  = cb_bf + (size_t)lvl * NUM_EMB * EMB_DIM;
    const float*          nrml = norms + lvl * NUM_EMB;

    for (int kt = 0; kt < NUM_EMB / 64; ++kt) {
      // async copy 64x128 bf16 codebook tile -> LDS (rows re-padded to RS)
      #pragma unroll
      for (int p = 0; p < 4; ++p) {
        int e   = t * 8 + p * 2048;                 // element offset in tile (8 bf16 = 16B unit)
        int row = e >> 7, col = e & 127;
        unsigned loff = lds_ebf + (unsigned)(row * RS + col) * 2u;
        const unsigned short* ga = cbl + (size_t)kt * 64 * EMB_DIM + e;
        asm volatile("global_load_async_to_lds_b128 %0, %1, off"
                     :: "v"(loff), "v"((uint64_t)(uintptr_t)ga) : "memory");
      }
      if (t < 64) en_s[t] = nrml[kt * 64 + t];
      asm volatile("s_wait_asynccnt 0" ::: "memory");
      __syncthreads();

      // WMMA: scores(64x64) = R_bf16 @ E_bf16^T, D=128 => 4 k-steps of 32
      const int h  = lane >> 4;
      const int rl = lane & 15;
      #pragma unroll
      for (int tt = 0; tt < 2; ++tt) {
        int tile = wave + tt * 8;                    // 16 output tiles over 8 waves
        int mi = tile >> 2, ni = tile & 3;
        v8f c = {0.f, 0.f, 0.f, 0.f, 0.f, 0.f, 0.f, 0.f};
        #pragma unroll
        for (int kc = 0; kc < 4; ++kc) {
          V16BF a, b;
          // A fragment: lane half h holds K = {h*8..h*8+7} and {16+h*8..}
          const uint4* ap = (const uint4*)&r_bf[(mi * 16 + rl) * RS + kc * 32 + h * 8];
          a.u[0] = ap[0];
          a.u[1] = ap[2];                            // +16 elems = +32B
          // B fragment (B = E^T): lane = column (code), half h holds K = {h*8..h*8+7} and {16+h*8..}
          const uint4* bp = (const uint4*)&e_bf[(ni * 16 + rl) * RS + kc * 32 + h * 8];
          b.u[0] = bp[0];
          b.u[1] = bp[2];
          c = __builtin_amdgcn_wmma_f32_16x16x32_bf16(false, a.v, false, b.v,
                                                      (short)0, c, false, false);
        }
        #pragma unroll
        for (int i = 0; i < 8; ++i) {
          int M = h * 8 + i;
          scores[(mi * 16 + M) * 65 + ni * 16 + rl] = c[i];
        }
      }
      __syncthreads();

      // argmin update over this 64-code tile (first-min tie break, ascending k)
      if (t < 64) {
        float bv = best[t]; int bi = bidx[t];
        #pragma unroll 8
        for (int k = 0; k < 64; ++k) {
          float d = en_s[k] - 2.0f * scores[t * 65 + k];
          if (d < bv) { bv = d; bi = kt * 64 + k; }
        }
        best[t] = bv; bidx[t] = bi;
      }
      __syncthreads();
    }

    // gather z_q (fp32, exact), losses, straight-through residual update
    {
      int idx = bidx[tok];
      const float4* eq = (const float4*)(cb + ((size_t)lvl * NUM_EMB + idx) * EMB_DIM + q * 32);
      #pragma unroll
      for (int i = 0; i < 8; ++i) {
        float4 v = eq[i];
        float d0 = v.x - rres[4*i+0], d1 = v.y - rres[4*i+1];
        float d2 = v.z - rres[4*i+2], d3 = v.w - rres[4*i+3];
        sqacc += d0*d0 + d1*d1 + d2*d2 + d3*d3;
        zqt[4*i+0] += v.x; zqt[4*i+1] += v.y; zqt[4*i+2] += v.z; zqt[4*i+3] += v.w;
        rres[4*i+0] -= v.x; rres[4*i+1] -= v.y; rres[4*i+2] -= v.z; rres[4*i+3] -= v.w;
      }
      if (q == 0) out_idx[(tokG0 + tok) * NUM_LEVELS + lvl] = (float)idx;
    }
    __syncthreads();   // protect r_bf / best / bidx before next level
  }

  // write z_q_total
  {
    float4* dst = (float4*)(out_zq + (tokG0 + tok) * EMB_DIM + q * 32);
    #pragma unroll
    for (int i = 0; i < 8; ++i) {
      float4 v; v.x = zqt[4*i+0]; v.y = zqt[4*i+1]; v.z = zqt[4*i+2]; v.w = zqt[4*i+3];
      dst[i] = v;
    }
  }

  // deterministic per-block loss partial
  red[t] = sqacc;
  __syncthreads();
  for (int s = 128; s > 0; s >>= 1) { if (t < s) red[t] += red[t + s]; __syncthreads(); }
  if (t == 0) partials[blockIdx.x] = red[0];
}

// ---------------- Kernel 2: deterministic final reduction -> scalars ----------------
__global__ void rvq_finalize(const float* __restrict__ partials, float* __restrict__ out) {
  __shared__ float red[256];
  const int t = threadIdx.x;
  float s = 0.f;
  for (int i = t; i < N_TOKENS / TM; i += 256) s += partials[i];  // fixed order
  red[t] = s;
  __syncthreads();
  for (int k = 128; k > 0; k >>= 1) { if (t < k) red[t] += red[t + k]; __syncthreads(); }
  if (t == 0) {
    float vq = red[0] / ((float)N_TOKENS * (float)EMB_DIM * (float)NUM_LEVELS);
    out[(size_t)N_TOKENS * EMB_DIM]     = vq;
    out[(size_t)N_TOKENS * EMB_DIM + 1] = 0.25f * vq;   // COMMITMENT_COST * same SSE
  }
}

extern "C" void kernel_launch(void* const* d_in, const int* in_sizes, int n_in,
                              void* d_out, int out_size, void* d_ws, size_t ws_size,
                              hipStream_t stream) {
  const float* z_e = (const float*)d_in[0];
  const float* cb  = (const float*)d_in[1];

  // workspace layout
  float*          partials = (float*)d_ws;                                   // 4096 * 4B
  float*          norms    = (float*)((char*)d_ws + 16384);                  // 3072 * 4B
  unsigned short* cb_bf    = (unsigned short*)((char*)d_ws + 32768);         // 786432 B

  float* out     = (float*)d_out;
  float* out_zq  = out;                                                      // [N, D]
  float* out_idx = out + (size_t)N_TOKENS * EMB_DIM + 2;                     // [N, 3] after 2 scalars

  rvq_prep<<<NUM_LEVELS * NUM_EMB, 128, 0, stream>>>(cb, cb_bf, norms);
  rvq_main<<<N_TOKENS / TM, 256, 0, stream>>>(z_e, cb, cb_bf, norms, out_zq, out_idx, partials);
  rvq_finalize<<<1, 256, 0, stream>>>(partials, out);
}
